// QuantLinear_62929860821185
// MI455X (gfx1250) — matrix operand order repaired and checked by reading, not compiled, not run on hardware
//
#include <hip/hip_runtime.h>

typedef _Float16 v2h  __attribute__((ext_vector_type(2)));
typedef _Float16 v4h  __attribute__((ext_vector_type(4)));
typedef _Float16 v8h  __attribute__((ext_vector_type(8)));
typedef _Float16 v16h __attribute__((ext_vector_type(16)));
typedef float    v4f  __attribute__((ext_vector_type(4)));
typedef float    v8f  __attribute__((ext_vector_type(8)));

#define K_DIM   4096
#define N_DIM   11008
#define NGROUPS 32      // K / 128
#define MT      256     // M tile per block
#define NT      128     // N tile per block
#define KT      32      // K per staged tile (one WMMA k-step)
#define KSTEPS  (K_DIM / KT)   // 128

// y[m,n] = sum_k x[m,k] * (((qw[k/8,n] >> 4*(k%8)) & 0xF) * s[n,k/128] - z[n,k/128]) + bias[n]
__global__ __launch_bounds__(256) void qlin4_wmma_f16(
    const float* __restrict__ x,       // (M, K) f32
    const int*   __restrict__ qw,      // (K/8, N) int32, 8 nibbles along K
    const float* __restrict__ scales,  // (N, 32) f32
    const float* __restrict__ zeros,   // (N, 32) f32
    const float* __restrict__ bias,    // (N,) f32
    float* __restrict__ out)           // (M, N) f32
{
    __shared__ _Float16 ldsA[2][MT * KT];  // [buf][m][k] row-major f16, 2 x 16 KB
    __shared__ _Float16 ldsB[2][NT * KT];  // [buf][n][k] n-major  f16, 2 x  8 KB

    const int tid  = threadIdx.x;       // 0..255 (8 waves of 32)
    const int lane = tid & 31;
    const int wave = tid >> 5;
    const int wm   = wave & 3;          // wave M slot: 0..3  (64 rows each)
    const int wn   = wave >> 2;         // wave N slot: 0..1  (64 cols each)
    const int l15  = lane & 15;
    const int lhi  = lane >> 4;         // 0 or 1

    const int m_base = blockIdx.y * MT;
    const int n_base = blockIdx.x * NT;

    v8f acc[4][4];                       // 4 M-frags x 4 N-frags, 128 VGPRs
#pragma unroll
    for (int f = 0; f < 4; ++f)
#pragma unroll
        for (int g = 0; g < 4; ++g)
#pragma unroll
            for (int e = 0; e < 8; ++e) acc[f][g][e] = 0.0f;

    // per-thread staging assignment (constant across K loop)
    const int n_loc  = tid & 127;           // B column handled by this thread
    const int kp0    = tid >> 7;            // 0..1 -> packed rows kp0, kp0+2
    const int n_glob = n_base + n_loc;

    // ---- pipelined staging registers ----
    v4f   areg[8];
    int   breg[2];
    float sv, zv;

    // issue global loads for K-step starting at k0 into registers
    auto load_step = [&](int k0) {
#pragma unroll
        for (int i = 0; i < 8; ++i) {
            const int l   = i * 1024 + tid * 4;   // 0..8191
            const int row = l >> 5;               // 0..255
            const int col = l & 31;               // multiple of 4
            areg[i] = *(const v4f*)(x + (size_t)(m_base + row) * K_DIM + k0 + col);
        }
#pragma unroll
        for (int j = 0; j < 2; ++j) {
            const int kp = kp0 + j * 2;           // 0..3
            breg[j] = qw[(size_t)((k0 >> 3) + kp) * N_DIM + n_glob];
        }
        const int grp = k0 >> 7;                  // group constant within 32-wide step
        sv = scales[(size_t)n_glob * NGROUPS + grp];
        zv = zeros [(size_t)n_glob * NGROUPS + grp];
    };

    // convert staged registers and store into LDS buffer `buf`
    auto store_step = [&](int buf) {
        _Float16* la = ldsA[buf];
        _Float16* lb = ldsB[buf];
#pragma unroll
        for (int i = 0; i < 8; ++i) {
            const int l   = i * 1024 + tid * 4;
            const int row = l >> 5;
            const int col = l & 31;
            v4h hv;
            hv[0] = (_Float16)areg[i][0];
            hv[1] = (_Float16)areg[i][1];
            hv[2] = (_Float16)areg[i][2];
            hv[3] = (_Float16)areg[i][3];
            *(v4h*)(la + row * KT + col) = hv;     // 8-byte aligned ds_store
        }
        // f16 magic-number dequant:
        //   0x6400|n is exactly f16(1024+n), so  n*s - z = (1024+n)*s + c,
        //   with c = -(1024*s + z), one v_pk_fma_f16 per nibble pair.
        const _Float16 hs = (_Float16)sv;
        const _Float16 hc = (_Float16)(-(1024.0f * sv + zv));
        const v2h hs2 = { hs, hs };
        const v2h hc2 = { hc, hc };
#pragma unroll
        for (int j = 0; j < 2; ++j) {
            const int kp = kp0 + j * 2;
            const unsigned q = (unsigned)breg[j];
            union { v8h h; v2h p[4]; } W;
#pragma unroll
            for (int p = 0; p < 4; ++p) {
                const unsigned b = q >> (8 * p);          // nibbles 2p, 2p+1 in [7:0]
                const unsigned pk = 0x64006400u | (b & 0xFu) | ((b & 0xF0u) << 12);
                union { unsigned u; v2h h; } t;
                t.u = pk;                                  // {1024+n_lo, 1024+n_hi}
                W.p[p] = t.h * hs2 + hc2;                  // v_pk_fma_f16
            }
            *(v8h*)(lb + n_loc * KT + kp * 8) = W.h;       // 16-byte aligned ds_store
        }
    };

    // ---- prologue: stage step 0 into buffer 0 ----
    load_step(0);
    store_step(0);

    for (int it = 0; it < KSTEPS; ++it) {
        __syncthreads();                 // buf[it&1] ready; buf[(it+1)&1] free
        const bool has_next = (it + 1 < KSTEPS);
        if (has_next) load_step((it + 1) * KT);   // global loads overlap compute

        // ---- compute from buf[it&1]: 4x4 fragment WMMAs ----
        const _Float16* la = ldsA[it & 1];
        const _Float16* lb = ldsB[it & 1];

        v16h bfr[4];
#pragma unroll
        for (int g = 0; g < 4; ++g) {
            const int coln = wn * 64 + g * 16 + l15;
            const int kb   = lhi * 16;   // lanes 0-15: K 0..15, lanes 16-31: K 16..31
            union { v16h v; v8h h[2]; } u;
            u.h[0] = *(const v8h*)(lb + coln * KT + kb);
            u.h[1] = *(const v8h*)(lb + coln * KT + kb + 8);
            bfr[g] = u.v;
        }
#pragma unroll
        for (int f = 0; f < 4; ++f) {
            const int rowm = wm * 64 + f * 16 + l15;
            const int ka   = lhi * 16;   // lanes 0-15: K 0..15, lanes 16-31: K 16..31
            union { v16h v; v8h h[2]; } u;
            u.h[0] = *(const v8h*)(la + rowm * KT + ka);
            u.h[1] = *(const v8h*)(la + rowm * KT + ka + 8);
            const v16h afr = u.v;
#pragma unroll
            for (int g = 0; g < 4; ++g) {
                acc[f][g] = __builtin_amdgcn_wmma_f32_16x16x32_f16(
                    false, afr, false, bfr[g], (short)0, acc[f][g], false, false);
            }
        }

        // ---- tail: convert + store next tile into the other buffer ----
        if (has_next) store_step((it + 1) & 1);
    }

    // ---- epilogue: C fragment layout -> global, + bias ----
    float bb[4];
#pragma unroll
    for (int g = 0; g < 4; ++g)
        bb[g] = bias[n_base + wn * 64 + g * 16 + l15];

#pragma unroll
    for (int f = 0; f < 4; ++f) {
        const int m0 = m_base + wm * 64 + f * 16 + lhi * 8;  // VGPR v holds M = m0 + v
#pragma unroll
        for (int g = 0; g < 4; ++g) {
            const int nc = n_base + wn * 64 + g * 16 + l15;
#pragma unroll
            for (int v = 0; v < 8; ++v) {
                out[(size_t)(m0 + v) * N_DIM + nc] = acc[f][g][v] + bb[g];
            }
        }
    }
}

extern "C" void kernel_launch(void* const* d_in, const int* in_sizes, int n_in,
                              void* d_out, int out_size, void* d_ws, size_t ws_size,
                              hipStream_t stream) {
    const float* x      = (const float*)d_in[0];
    const int*   qw     = (const int*)  d_in[1];
    const float* scales = (const float*)d_in[2];
    const float* zeros  = (const float*)d_in[3];
    const float* bias   = (const float*)d_in[4];
    float* out = (float*)d_out;

    const int M = in_sizes[0] / K_DIM;   // 2*2048 = 4096
    dim3 grid(N_DIM / NT, M / MT);       // (86, 16)
    dim3 block(256);
    qlin4_wmma_f16<<<grid, block, 0, stream>>>(x, qw, scales, zeros, bias, out);
}
